// MLP_12953621364900
// MI455X (gfx1250) — matrix operand order repaired and checked by reading, not compiled, not run on hardware
//
#include <hip/hip_runtime.h>

// ---------------------------------------------------------------------------
// Problem constants (T,B,C,H,W) = (4,64,384,14,14), NC=2
// ---------------------------------------------------------------------------
#define TT   4
#define BB   64
#define CC   384
#define HW   196          // 14*14
#define O2   768          // NC*C
#define SL   (TT*BB)      // 256 (t,b) slices

typedef __bf16 bf16_t;
typedef __bf16 v16bf __attribute__((ext_vector_type(16)));
typedef float  v8f   __attribute__((ext_vector_type(8)));

union FragA { uint4 u[2]; v16bf v; };
union FragB { uint4 u[2]; v16bf v; };

__device__ __forceinline__ bf16_t ld_bf16(const float* p)  { return (bf16_t)*p; }
__device__ __forceinline__ bf16_t ld_bf16(const bf16_t* p) { return *p; }

// ---------------------------------------------------------------------------
// fp32 -> bf16 weight conversion
// ---------------------------------------------------------------------------
__global__ __launch_bounds__(256) void cvt_f32_bf16(const float* __restrict__ in,
                                                    bf16_t* __restrict__ out, int n) {
    int i = blockIdx.x * 256 + threadIdx.x;
    if (i < n) out[i] = (bf16_t)in[i];
}

// ---------------------------------------------------------------------------
// Per-slice GEMM with bias:  Out[s,o,hw] = sum_c W[o,c] * X[s,c,hw] + bias[o]
//   W   : O x K  (bf16, row-major)
//   X   : SL x K x HW  (TIN = float or bf16), each slice contiguous K x HW
//   Out : SL x O x HW  (fp32)
// Block = 256 threads = 8 waves; block tile = 128(M) x 64(N);
// wave tile = 64(M) x 16(N) via 4 accumulators; K stepped by 32.
// grid = (ceil(HW/64), O/128, SL)
// ---------------------------------------------------------------------------
template <typename TIN>
__global__ __launch_bounds__(256) void gemm_bias(const bf16_t* __restrict__ W,
                                                 const TIN* __restrict__ X,
                                                 const float* __restrict__ bias,
                                                 float* __restrict__ Out,
                                                 int O, int K) {
    const int lane  = threadIdx.x & 31;
    const int wave  = threadIdx.x >> 5;
    const int wm    = wave >> 2;            // 0..1
    const int wn    = wave & 3;             // 0..3
    const int slice = blockIdx.z;

    const int m_base  = (blockIdx.y * 2 + wm) * 64;
    const int n_base  = blockIdx.x * 64 + wn * 16;

    const int mrow    = lane & 15;          // A row / D col within tile
    const int khalf   = lane >> 4;          // 0/1
    const int row_off = khalf * 8;          // D row offset for this lane half

    const int  ncol   = n_base + mrow;
    const bool nok    = ncol < HW;
    const int  nclamp = nok ? ncol : (HW - 1);

    const TIN* Xs = X + (size_t)slice * K * HW;

    // accumulators initialized with bias (per output channel = D row)
    v8f acc[4];
#pragma unroll
    for (int a = 0; a < 4; ++a) {
#pragma unroll
        for (int i = 0; i < 8; ++i)
            acc[a][i] = bias[m_base + a * 16 + row_off + i];
    }

    for (int k0 = 0; k0 < K; k0 += 32) {
        // ---- B fragment: col n = lane&15, K = khalf*16 + j (j = 0..15) ----
        FragB bf;
        const TIN* bp = Xs + (size_t)(k0 + khalf * 16) * HW + nclamp;
#pragma unroll
        for (int j = 0; j < 16; ++j) {
            bf16_t e = ld_bf16(bp + (size_t)j * HW);
            bf.v[j] = nok ? e : (bf16_t)0.0f;
        }
        // ---- A fragments (2x16B per lane) + WMMA ----
#pragma unroll
        for (int a = 0; a < 4; ++a) {
            const bf16_t* Ar = W + (size_t)(m_base + a * 16 + mrow) * K
                                 + k0 + khalf * 16;
            FragA af;
            af.u[0] = *(const uint4*)(Ar);        // K = k0 + khalf*16 + 0..7
            af.u[1] = *(const uint4*)(Ar + 8);    // K = k0 + khalf*16 + 8..15
            acc[a] = __builtin_amdgcn_wmma_f32_16x16x32_bf16(
                false, af.v, false, bf.v, (short)0, acc[a], false, false);
        }
    }

    // ---- store D (guard padded N columns) ----
    if (nok) {
        float* op = Out + (size_t)slice * O * HW + ncol;
#pragma unroll
        for (int a = 0; a < 4; ++a) {
#pragma unroll
            for (int i = 0; i < 8; ++i)
                op[(size_t)(m_base + a * 16 + row_off + i) * HW] = acc[a][i];
        }
    }
}

// ---------------------------------------------------------------------------
// Per-channel BN statistics over (T,B,H,W): one block per channel.
// Emits (scale, shift): y = h*scale + shift
// ---------------------------------------------------------------------------
__global__ __launch_bounds__(256) void bn_stats(const float* __restrict__ h,
                                                const float* __restrict__ gamma,
                                                const float* __restrict__ beta,
                                                float2* __restrict__ ss, int Cn) {
    const int c   = blockIdx.x;
    const int tid = threadIdx.x;
    const int Nper = SL * HW;           // 50176 elements per channel

    float sum = 0.f, sq = 0.f;
    for (int j = tid; j < Nper; j += 256) {
        int s  = j / HW;
        int hw = j - s * HW;
        float v = h[((size_t)s * Cn + c) * HW + hw];
        sum += v;
        sq  += v * v;
    }
    __shared__ float rs[256], rq[256];
    rs[tid] = sum; rq[tid] = sq;
    __syncthreads();
    for (int st = 128; st > 0; st >>= 1) {
        if (tid < st) { rs[tid] += rs[tid + st]; rq[tid] += rq[tid + st]; }
        __syncthreads();
    }
    if (tid == 0) {
        float inv   = 1.0f / (float)Nper;
        float mean  = rs[0] * inv;
        float var   = rq[0] * inv - mean * mean;   // biased var, matches jnp.var
        float scale = gamma[c] * rsqrtf(var + 1e-5f);
        ss[c] = make_float2(scale, beta[c] - mean * scale);
    }
}

// ---------------------------------------------------------------------------
// Fused BN + 4-step LIF.  v' = (v + y)/2 ; spike = v' >= 1 ; v = spike ? 0 : v'
// One thread per (b,c,hw) element; sequential over T in registers.
// ---------------------------------------------------------------------------
template <typename TOUT>
__global__ __launch_bounds__(256) void bn_lif(const float* __restrict__ h,
                                              const float2* __restrict__ ss,
                                              TOUT* __restrict__ out,
                                              int Cn, int per_t) {
    size_t e = (size_t)blockIdx.x * 256 + threadIdx.x;
    if (e >= (size_t)per_t) return;
    int c = (int)((e / HW) % Cn);
    float2 s2 = ss[c];
    float v = 0.0f;
#pragma unroll
    for (int t = 0; t < TT; ++t) {
        size_t idx = (size_t)t * per_t + e;
        float y = h[idx] * s2.x + s2.y;
        v = 0.5f * (v + y);
        bool sp = (v >= 1.0f);
        out[idx] = (TOUT)(sp ? 1.0f : 0.0f);
        v = sp ? 0.0f : v;
    }
}

// ---------------------------------------------------------------------------
// Launch: w->bf16 | GEMM1 | stats1 | BN+LIF1 (bf16 spikes) |
//         GEMM2 | stats2 | BN+LIF2 -> d_out
// ---------------------------------------------------------------------------
extern "C" void kernel_launch(void* const* d_in, const int* in_sizes, int n_in,
                              void* d_out, int out_size, void* d_ws, size_t ws_size,
                              hipStream_t stream) {
    const float* x     = (const float*)d_in[0];
    const float* w1    = (const float*)d_in[1];
    const float* b1    = (const float*)d_in[2];
    const float* g1    = (const float*)d_in[3];
    const float* beta1 = (const float*)d_in[4];
    const float* w2    = (const float*)d_in[5];
    const float* b2    = (const float*)d_in[6];
    const float* g2    = (const float*)d_in[7];
    const float* beta2 = (const float*)d_in[8];
    float* out = (float*)d_out;

    char* ws = (char*)d_ws;
    size_t off = 0;
    auto take = [&](size_t bytes) -> char* {
        char* p = ws + off;
        off += (bytes + 255) & ~(size_t)255;
        return p;
    };
    bf16_t* w1b = (bf16_t*)take((size_t)CC * CC * sizeof(bf16_t));
    bf16_t* w2b = (bf16_t*)take((size_t)O2 * CC * sizeof(bf16_t));
    float2* ss1 = (float2*)take((size_t)CC * sizeof(float2));
    float2* ss2 = (float2*)take((size_t)O2 * sizeof(float2));
    float*  h1  = (float*) take((size_t)SL * CC * HW * sizeof(float));
    bf16_t* spk = (bf16_t*)take((size_t)SL * CC * HW * sizeof(bf16_t));
    float*  h2  = (float*) take((size_t)SL * O2 * HW * sizeof(float));
    (void)ws_size;

    // 1. weights -> bf16
    cvt_f32_bf16<<<(CC * CC + 255) / 256, 256, 0, stream>>>(w1, w1b, CC * CC);
    cvt_f32_bf16<<<(O2 * CC + 255) / 256, 256, 0, stream>>>(w2, w2b, O2 * CC);

    // 2. GEMM1: h1 = w1 * x + b1   (fp32 input, bf16 WMMA, fp32 acc)
    {
        dim3 grid((HW + 63) / 64, CC / 128, SL);
        gemm_bias<float><<<grid, 256, 0, stream>>>(w1b, x, b1, h1, CC, CC);
    }
    // 3. BN stats 1
    bn_stats<<<CC, 256, 0, stream>>>(h1, g1, beta1, ss1, CC);

    // 4. BN + LIF1 -> bf16 spikes (exactly 0/1 in bf16)
    {
        int per_t = BB * CC * HW;
        bn_lif<bf16_t><<<(per_t + 255) / 256, 256, 0, stream>>>(h1, ss1, spk, CC, per_t);
    }
    // 5. GEMM2: h2 = w2 * spk + b2
    {
        dim3 grid((HW + 63) / 64, O2 / 128, SL);
        gemm_bias<bf16_t><<<grid, 256, 0, stream>>>(w2b, spk, b2, h2, O2, CC);
    }
    // 6. BN stats 2
    bn_stats<<<O2, 256, 0, stream>>>(h2, g2, beta2, ss2, O2);

    // 7. BN + LIF2 -> final fp32 spikes
    {
        int per_t = BB * O2 * HW;
        bn_lif<float><<<(per_t + 255) / 256, 256, 0, stream>>>(h2, ss2, out, O2, per_t);
    }
}
